// MegablockMoE_343597384324
// MI455X (gfx1250) — matrix operand order repaired and checked by reading, not compiled, not run on hardware
//
#include <hip/hip_runtime.h>

// ---------------------------------------------------------------------------
// MoE (megablocks-style) for MI455X / gfx1250, wave32 + WMMA bf16.
// Problem constants (match reference): B=4,S=2048,D=1024,F=4096,E=8,K=2
// T = 8192 tokens, capacity C = K*T/E = 2048.
//
// Key layout decision: weight bf16 copies are stored TRANSPOSED
// (w1t[E][F][D], w2t[E][D][F]) so WMMA B-fragments (one column, 16
// K values per lane, two runs of 8) are contiguous in global memory -> B-tile
// staging is identical to A-tile staging (b128 loads, b64 LDS stores,
// conflict-free b128 LDS fragment reads). No scalar LDS transpose scatter
// in the GEMM inner loop.
// ---------------------------------------------------------------------------

typedef unsigned short u16;
typedef __attribute__((ext_vector_type(16))) __bf16 v16bf;
typedef __attribute__((ext_vector_type(8)))  float  v8f;

union Frag16 { uint4 u[2]; v16bf v; };   // 16 bf16 = 8 VGPRs

#define NE     8
#define TOPK   2
#define DD     1024
#define FF     4096
#define TT     8192
#define CC     2048
#define NASSIGN (TT*TOPK)        // 16384
#define NBLK    (NASSIGN/256)    // 64

#define BM   128
#define BN   128
#define BK   32
#define LSTR 40                  // BK + 8 halves padding -> conflict-free b128 LDS reads

__device__ __forceinline__ u16 f2bf(float f) {
  unsigned int u = __float_as_uint(f);
  unsigned int r = u + 0x7FFFu + ((u >> 16) & 1u);   // round-to-nearest-even
  return (u16)(r >> 16);
}

// ---------------------------------------------------------------------------
// fused transpose + fp32->bf16 for weights:
//   src: [E][R][Cdim] f32 (row-major), dst: [E][Cdim][R] bf16
// 32x32 tiles through LDS; coalesced reads and writes.
// grid: (Cdim/32, R/32, E), 256 threads
// ---------------------------------------------------------------------------
__global__ __launch_bounds__(256)
void transpose_bf16_kernel(const float* __restrict__ src, u16* __restrict__ dst,
                           int R, int Cdim) {
  __shared__ float tile[32][33];
  const int tid = threadIdx.x;
  const int c0 = blockIdx.x * 32;
  const int r0 = blockIdx.y * 32;
  const size_t eoff = (size_t)blockIdx.z * (size_t)R * (size_t)Cdim;

  const int rr = tid >> 3;          // 0..31
  const int q  = tid & 7;           // 0..7 -> 4 floats each

  const float4 v = *(const float4*)&src[eoff + (size_t)(r0 + rr) * Cdim + c0 + q * 4];
  tile[rr][q * 4 + 0] = v.x;
  tile[rr][q * 4 + 1] = v.y;
  tile[rr][q * 4 + 2] = v.z;
  tile[rr][q * 4 + 3] = v.w;
  __syncthreads();

  // dst row = (c0 + rr) over Cdim, columns (r0 + q*4 .. +3) over R
  const float t0 = tile[q * 4 + 0][rr];
  const float t1 = tile[q * 4 + 1][rr];
  const float t2 = tile[q * 4 + 2][rr];
  const float t3 = tile[q * 4 + 3][rr];
  uint2 p;
  p.x = (unsigned)f2bf(t0) | ((unsigned)f2bf(t1) << 16);
  p.y = (unsigned)f2bf(t2) | ((unsigned)f2bf(t3) << 16);
  *(uint2*)&dst[eoff + (size_t)(c0 + rr) * R + r0 + q * 4] = p;
}

// ---------------------------------------------------------------------------
// RMSNorm + router (softmax over 8 experts, top-2, no renorm)
// one 256-thread block per token; each thread owns 4 channels of D=1024
// ---------------------------------------------------------------------------
__global__ __launch_bounds__(256)
void rms_router_kernel(const float* __restrict__ x, const float* __restrict__ rw,
                       const float* __restrict__ wr, u16* __restrict__ tn,
                       int* __restrict__ eidx, float* __restrict__ gate) {
  __shared__ float red[256];
  __shared__ float logits[NE];
  const int t = blockIdx.x, tid = threadIdx.x;
  const int d = tid * 4;

  const float4 v = *(const float4*)&x[(size_t)t * DD + d];
  red[tid] = v.x * v.x + v.y * v.y + v.z * v.z + v.w * v.w;
  __syncthreads();
  for (int s = 128; s > 0; s >>= 1) { if (tid < s) red[tid] += red[tid + s]; __syncthreads(); }
  const float rms = rsqrtf(red[0] * (1.0f / DD) + 1e-6f);
  __syncthreads();

  const float4 w4 = *(const float4*)&rw[d];
  float h[4] = { v.x * rms * w4.x, v.y * rms * w4.y, v.z * rms * w4.z, v.w * rms * w4.w };

  uint2 p;
  p.x = (unsigned)f2bf(h[0]) | ((unsigned)f2bf(h[1]) << 16);
  p.y = (unsigned)f2bf(h[2]) | ((unsigned)f2bf(h[3]) << 16);
  *(uint2*)&tn[(size_t)t * DD + d] = p;

  float pl[NE];
  #pragma unroll
  for (int e = 0; e < NE; ++e) pl[e] = 0.0f;
  #pragma unroll
  for (int j = 0; j < 4; ++j) {
    const float* wrow = &wr[(size_t)(d + j) * NE];
    #pragma unroll
    for (int e = 0; e < NE; ++e) pl[e] += h[j] * wrow[e];
  }
  for (int e = 0; e < NE; ++e) {
    red[tid] = pl[e]; __syncthreads();
    for (int s = 128; s > 0; s >>= 1) { if (tid < s) red[tid] += red[tid + s]; __syncthreads(); }
    if (tid == 0) logits[e] = red[0];
    __syncthreads();
  }

  if (tid == 0) {
    float m = logits[0];
    #pragma unroll
    for (int e = 1; e < NE; ++e) m = fmaxf(m, logits[e]);
    float ex[NE], s = 0.0f;
    #pragma unroll
    for (int e = 0; e < NE; ++e) { ex[e] = __expf(logits[e] - m); s += ex[e]; }
    const float inv = 1.0f / s;
    int b0 = 0; float v0 = -1.0f;
    #pragma unroll
    for (int e = 0; e < NE; ++e) { float sc = ex[e] * inv; if (sc > v0) { v0 = sc; b0 = e; } }
    int b1 = 0; float v1 = -1.0f;
    #pragma unroll
    for (int e = 0; e < NE; ++e) { if (e == b0) continue; float sc = ex[e] * inv; if (sc > v1) { v1 = sc; b1 = e; } }
    eidx[t * TOPK + 0] = b0; eidx[t * TOPK + 1] = b1;
    gate[t * TOPK + 0] = v0; gate[t * TOPK + 1] = v1;
  }
}

// ---------------------------------------------------------------------------
// token-major capacity assignment (deterministic): histogram / scan / rank
// ---------------------------------------------------------------------------
__global__ __launch_bounds__(256)
void count_kernel(const int* __restrict__ eidx, int* __restrict__ bcnt) {
  __shared__ int c[NE];
  const int tid = threadIdx.x;
  if (tid < NE) c[tid] = 0;
  __syncthreads();
  atomicAdd(&c[eidx[blockIdx.x * 256 + tid]], 1);
  __syncthreads();
  if (tid < NE) bcnt[blockIdx.x * NE + tid] = c[tid];
}

__global__ void scan_kernel(const int* __restrict__ bcnt, int* __restrict__ boff) {
  const int e = threadIdx.x;
  if (e >= NE) return;
  int run = 0;
  for (int b = 0; b < NBLK; ++b) { boff[b * NE + e] = run; run += bcnt[b * NE + e]; }
}

__global__ __launch_bounds__(256)
void assign_kernel(const int* __restrict__ eidx, const float* __restrict__ gate,
                   const int* __restrict__ boff, int* __restrict__ slot,
                   float* __restrict__ gk) {
  __shared__ unsigned char se[256];
  const int tid = threadIdx.x;
  const int i = blockIdx.x * 256 + tid;
  const int e = eidx[i];
  se[tid] = (unsigned char)e;
  __syncthreads();
  int rank = 0;
  for (int j = 0; j < tid; ++j) rank += (se[j] == (unsigned char)e) ? 1 : 0;
  const int pos = boff[blockIdx.x * NE + e] + rank;
  const bool keep = pos < CC;
  slot[i] = keep ? (e * CC + pos) : -1;
  gk[i]   = keep ? gate[i] : 0.0f;
}

// ---------------------------------------------------------------------------
// dispatch: zero x_buf then scatter kept bf16 rows (slots are unique)
// ---------------------------------------------------------------------------
__global__ __launch_bounds__(256)
void zero_kernel(uint4* __restrict__ p, long long n4) {
  long long i = (long long)blockIdx.x * 256 + threadIdx.x;
  const long long stride = (long long)gridDim.x * 256;
  const uint4 z = make_uint4(0u, 0u, 0u, 0u);
  for (; i < n4; i += stride) p[i] = z;
}

__global__ __launch_bounds__(256)
void dispatch_kernel(const u16* __restrict__ tn, const int* __restrict__ slot,
                     u16* __restrict__ xb) {
  const int i = blockIdx.x;
  const int s = slot[i];
  if (s < 0) return;
  const int tok = i >> 1;                       // i = tok*TOPK + k
  const uint2* src = (const uint2*)&tn[(size_t)tok * DD];
  uint2* dst = (uint2*)&xb[(size_t)s * DD];
  dst[threadIdx.x] = src[threadIdx.x];          // 256 * 4 bf16 = 1024
}

// ---------------------------------------------------------------------------
// WMMA bf16 grouped GEMM:  C[z] = act( A[z](CC x KDIM) @ Bt[z]^T )
//   A : [CC][KDIM]  bf16 row-major (K contiguous)
//   Bt: [NDIM][KDIM] bf16 row-major (K contiguous) == B transposed
// 256 threads = 8 waves; block tile 128x128, K-step 32; wave tile 32x64
// GELU_BF16OUT: fuse exact GELU and store bf16, else store f32.
// ---------------------------------------------------------------------------
template<int KDIM, int NDIM, bool GELU_BF16OUT>
__global__ __launch_bounds__(256)
void wmma_gemm(const u16* __restrict__ A, const u16* __restrict__ Bt,
               void* __restrict__ Cout) {
  __shared__ u16 sA[2][BM * LSTR];
  __shared__ u16 sB[2][BN * LSTR];   // [n][k], K-contiguous (same shape as sA)

  const int tid   = threadIdx.x;
  const int lane  = tid & 31;
  const int half_ = lane >> 4;       // K-half selector per ISA 16-bit layouts
  const int lrow  = lane & 15;
  const int wv    = tid >> 5;        // wave 0..7
  const int wm    = (wv & 3) * 32;   // 4 waves along M
  const int wn    = (wv >> 2) * 64;  // 2 waves along N

  const int bm0 = blockIdx.y * BM;
  const int bn0 = blockIdx.x * BN;
  const u16* Ae = A  + (size_t)blockIdx.z * (size_t)CC   * KDIM;
  const u16* Be = Bt + (size_t)blockIdx.z * (size_t)NDIM * KDIM;
  const size_t ec = (size_t)blockIdx.z * (size_t)CC * NDIM;

  // staging index map (global -> reg -> LDS); identical for A and B tiles
  const int srow = tid >> 2, sq = tid & 3;   // 2 its x (row srow/srow+64, 8 halves at sq*8)

  v8f acc[2][4];
  {
    const v8f z = {0.f, 0.f, 0.f, 0.f, 0.f, 0.f, 0.f, 0.f};
    #pragma unroll
    for (int i = 0; i < 2; ++i)
      #pragma unroll
      for (int j = 0; j < 4; ++j) acc[i][j] = z;
  }

  uint4 ra[2], rb[2];

  auto gload = [&](int kt) {
    const int k0 = kt * BK;
    #pragma unroll
    for (int it = 0; it < 2; ++it)
      ra[it] = *(const uint4*)&Ae[(size_t)(bm0 + srow + it * 64) * KDIM + k0 + sq * 8];
    #pragma unroll
    for (int it = 0; it < 2; ++it)
      rb[it] = *(const uint4*)&Be[(size_t)(bn0 + srow + it * 64) * KDIM + k0 + sq * 8];
  };

  auto lstore = [&](int buf) {
    #pragma unroll
    for (int it = 0; it < 2; ++it) {
      u16* d = &sA[buf][(srow + it * 64) * LSTR + sq * 8];
      *(uint2*)(d + 0) = make_uint2(ra[it].x, ra[it].y);
      *(uint2*)(d + 4) = make_uint2(ra[it].z, ra[it].w);
    }
    #pragma unroll
    for (int it = 0; it < 2; ++it) {
      u16* d = &sB[buf][(srow + it * 64) * LSTR + sq * 8];
      *(uint2*)(d + 0) = make_uint2(rb[it].x, rb[it].y);
      *(uint2*)(d + 4) = make_uint2(rb[it].z, rb[it].w);
    }
  };

  gload(0);
  lstore(0);
  __syncthreads();

  const int KT = KDIM / BK;
  for (int kt = 0; kt < KT; ++kt) {
    const int cur = kt & 1;
    if (kt + 1 < KT) gload(kt + 1);
    if (kt + 2 < KT) {     // stream the k+2 tile into cache (global_prefetch_b8)
      __builtin_prefetch(&Ae[(size_t)(bm0 + srow) * KDIM + (kt + 2) * BK + sq * 8], 0, 1);
      __builtin_prefetch(&Be[(size_t)(bn0 + srow) * KDIM + (kt + 2) * BK + sq * 8], 0, 1);
    }

    Frag16 af[2], bfr[4];
    #pragma unroll
    for (int i = 0; i < 2; ++i) {   // A 16x32: lane-half K split {0..7,16..23}/{8..15,24..31}
      const u16* p = &sA[cur][(wm + i * 16 + lrow) * LSTR + half_ * 8];
      af[i].u[0] = *(const uint4*)p;
      af[i].u[1] = *(const uint4*)(p + 16);
    }
    #pragma unroll
    for (int j = 0; j < 4; ++j) {   // B 32x16: column lrow, lane-half K split {0..7,16..23}/{8..15,24..31}
      const u16* p = &sB[cur][(wn + j * 16 + lrow) * LSTR + half_ * 8];
      bfr[j].u[0] = *(const uint4*)p;
      bfr[j].u[1] = *(const uint4*)(p + 16);
    }

    #pragma unroll
    for (int i = 0; i < 2; ++i)
      #pragma unroll
      for (int j = 0; j < 4; ++j)
        acc[i][j] = __builtin_amdgcn_wmma_f32_16x16x32_bf16(
            false, af[i].v, false, bfr[j].v, (short)0, acc[i][j], false, false);

    if (kt + 1 < KT) lstore(cur ^ 1);
    __syncthreads();
  }

  // epilogue: C/D layout — lane<16: M=r, lane>=16: M=8+r, N=lane%16
  #pragma unroll
  for (int i = 0; i < 2; ++i) {
    #pragma unroll
    for (int j = 0; j < 4; ++j) {
      const int col = bn0 + wn + j * 16 + lrow;
      #pragma unroll
      for (int r = 0; r < 8; ++r) {
        const int row = bm0 + wm + i * 16 + half_ * 8 + r;
        float v = acc[i][j][r];
        if constexpr (GELU_BF16OUT) {
          v = 0.5f * v * (1.0f + erff(v * 0.70710678118654752f));   // exact gelu
          ((u16*)Cout)[ec + (size_t)row * NDIM + col] = f2bf(v);
        } else {
          ((float*)Cout)[ec + (size_t)row * NDIM + col] = v;
        }
      }
    }
  }
}

// ---------------------------------------------------------------------------
// combine: out[t] = sum_k gate_k * y_buf[slot_k]
// ---------------------------------------------------------------------------
__global__ __launch_bounds__(256)
void combine_kernel(const float* __restrict__ yb, const int* __restrict__ slot,
                    const float* __restrict__ gk, float* __restrict__ out) {
  const int t = blockIdx.x;
  const int d = threadIdx.x * 4;
  float4 acc = make_float4(0.f, 0.f, 0.f, 0.f);
  #pragma unroll
  for (int k = 0; k < TOPK; ++k) {
    const int i = t * TOPK + k;
    const int s = slot[i];
    if (s >= 0) {
      const float g = gk[i];
      const float4 yv = *(const float4*)&yb[(size_t)s * DD + d];
      acc.x += g * yv.x; acc.y += g * yv.y; acc.z += g * yv.z; acc.w += g * yv.w;
    }
  }
  *(float4*)&out[(size_t)t * DD + d] = acc;
}

// ---------------------------------------------------------------------------
extern "C" void kernel_launch(void* const* d_in, const int* in_sizes, int n_in,
                              void* d_out, int out_size, void* d_ws, size_t ws_size,
                              hipStream_t stream) {
  (void)in_sizes; (void)n_in; (void)out_size; (void)ws_size;
  const float* x  = (const float*)d_in[0];
  const float* rw = (const float*)d_in[1];
  const float* wr = (const float*)d_in[2];
  const float* w1 = (const float*)d_in[3];
  const float* w2 = (const float*)d_in[4];
  float* out = (float*)d_out;

  char* ws = (char*)d_ws;
  size_t off = 0;
  auto carve = [&](size_t bytes) -> void* {
    void* p = ws + off;
    off += (bytes + 255) & ~(size_t)255;
    return p;
  };
  u16*   tn   = (u16*)  carve((size_t)TT * DD * 2);        //  16 MB normalized tokens (bf16)
  u16*   w1t  = (u16*)  carve((size_t)NE * DD * FF * 2);   //  64 MB w1^T bf16 [E][F][D]
  u16*   w2t  = (u16*)  carve((size_t)NE * FF * DD * 2);   //  64 MB w2^T bf16 [E][D][F]
  u16*   xb   = (u16*)  carve((size_t)NE * CC * DD * 2);   //  32 MB dispatched tokens
  u16*   mid  = (u16*)  carve((size_t)NE * CC * FF * 2);   // 128 MB gelu(x@w1) bf16
  float* yb   = (float*)carve((size_t)NE * CC * DD * 4);   //  64 MB expert outputs
  int*   eidx = (int*)  carve((size_t)NASSIGN * 4);
  float* gate = (float*)carve((size_t)NASSIGN * 4);
  int*   slot = (int*)  carve((size_t)NASSIGN * 4);
  float* gk   = (float*)carve((size_t)NASSIGN * 4);
  int*   bcnt = (int*)  carve((size_t)NBLK * NE * 4);
  int*   boff = (int*)  carve((size_t)NBLK * NE * 4);

  // 1) weights -> transposed bf16 (L2-resident afterwards: 128 MB << 192 MB L2)
  transpose_bf16_kernel<<<dim3(FF / 32, DD / 32, NE), 256, 0, stream>>>(w1, w1t, DD, FF);
  transpose_bf16_kernel<<<dim3(DD / 32, FF / 32, NE), 256, 0, stream>>>(w2, w2t, FF, DD);

  // 2) RMSNorm + router (top-2, softmax gates)
  rms_router_kernel<<<TT, 256, 0, stream>>>(x, rw, wr, tn, eidx, gate);

  // 3) deterministic token-major capacity assignment
  count_kernel<<<NBLK, 256, 0, stream>>>(eidx, bcnt);
  scan_kernel<<<1, 32, 0, stream>>>(bcnt, boff);
  assign_kernel<<<NBLK, 256, 0, stream>>>(eidx, gate, boff, slot, gk);

  // 4) dispatch into [E][C][D] bf16 buffers
  zero_kernel<<<2048, 256, 0, stream>>>((uint4*)xb, (long long)((size_t)NE * CC * DD * 2 / 16));
  dispatch_kernel<<<NASSIGN, 256, 0, stream>>>(tn, slot, xb);

  // 5) grouped expert FFN on the WMMA path
  //    GEMM1: mid = gelu(xb @ w1)   A=[CC][DD], Bt=w1t[FF][DD], out bf16 [CC][FF]
  //    GEMM2: yb  = mid @ w2        A=[CC][FF], Bt=w2t[DD][FF], out f32  [CC][DD]
  wmma_gemm<DD, FF, true ><<<dim3(FF / BN, CC / BM, NE), 256, 0, stream>>>(xb,  w1t, (void*)mid);
  wmma_gemm<FF, DD, false><<<dim3(DD / BN, CC / BM, NE), 256, 0, stream>>>(mid, w2t, (void*)yb);

  // 6) gather + gate + sum
  combine_kernel<<<TT, 256, 0, stream>>>(yb, slot, gk, out);
}
